// AMEncoder_3384434229672
// MI455X (gfx1250) — matrix operand order repaired and checked by reading, not compiled, not run on hardware
//
#include <hip/hip_runtime.h>
#include <hip/hip_bf16.h>

// ---------------- model dims ----------------
#define D_   512
#define H_   8
#define DH_  64
#define L_   6
#define FF_  2048
#define V_   4096
#define B_   2
#define S_   4096
#define T_   (S_*B_)      // 8192 tokens
#define EPS_ 1e-5f

// ---------------- WMMA types ----------------
typedef __attribute__((ext_vector_type(16))) __bf16 v16bf;
typedef __attribute__((ext_vector_type(8)))  __bf16 v8bf;
typedef __attribute__((ext_vector_type(8)))  float  v8f;

typedef unsigned int u32x4 __attribute__((ext_vector_type(4)));
typedef int          i32x8 __attribute__((ext_vector_type(8)));
typedef int          i32x4 __attribute__((ext_vector_type(4)));

#if defined(__AMDGCN__) && __has_builtin(__builtin_amdgcn_tensor_load_to_lds) && __has_builtin(__builtin_amdgcn_s_wait_tensorcnt)
#define USE_TDM 1
#else
#define USE_TDM 0
#endif

static __device__ inline v8f zero8() {
    v8f z = {0.f,0.f,0.f,0.f,0.f,0.f,0.f,0.f};
    return z;
}

// A-fragment (16x32, bf16): lane l (0..15) row M=l holds K = {kb..kb+7, kb+16..kb+23},
// lanes 16..31 same rows with kb=8.  base points at [row0, k0], ld = row stride (elems).
static __device__ inline v16bf frag_a(const __bf16* base, int ld) {
    int l   = threadIdx.x & 31;
    int row = l & 15;
    int kb  = (l >> 4) * 8;
    const v8bf* p0 = (const v8bf*)(base + (size_t)row * ld + kb);
    const v8bf* p1 = (const v8bf*)(base + (size_t)row * ld + kb + 16);
    v8bf lo = *p0, hi = *p1;
    return __builtin_shufflevector(lo, hi, 0,1,2,3,4,5,6,7,8,9,10,11,12,13,14,15);
}

// B-fragment (32x16, bf16): lane n (0..15) holds column n, K = {kb..kb+7, kb+16..kb+23};
// lanes 16..31 same columns with kb=8.
static __device__ inline v16bf frag_b(const __bf16* base, int ld) {
    int l  = threadIdx.x & 31;
    int n  = l & 15;
    int kb = (l >> 4) * 8;
    const v8bf* p0 = (const v8bf*)(base + (size_t)n * ld + kb);
    const v8bf* p1 = (const v8bf*)(base + (size_t)n * ld + kb + 16);
    v8bf lo = *p0, hi = *p1;
    return __builtin_shufflevector(lo, hi, 0,1,2,3,4,5,6,7,8,9,10,11,12,13,14,15);
}

static __device__ inline v8f wmma_bf16(v16bf a, v16bf b, v8f c) {
    return __builtin_amdgcn_wmma_f32_16x16x32_bf16(
        /*neg_a=*/false, a, /*neg_b=*/false, b,
        /*c_mod=*/(short)0, c, /*reuse_a=*/false, /*reuse_b=*/false);
}

// ---------------- generic big GEMM:  out[M,N] = act[M,K] * W[N,K]^T + bias ----------------
#define TM 128
#define TN 128
#define TK 32
#define LDSS 40   // padded row stride (elems); 80B rows keep 16B alignment

enum { EPI_F32 = 0, EPI_RELU_BF16 = 1, EPI_RES = 2, EPI_LOGITS = 3 };

#if USE_TDM
// Issue one TDM 2-D tile load: tile = 128 rows x TK elems (bf16), row stride = ld elems.
// LDS destination rows are padded: every 64 bytes (pad_interval=3) insert 4 DWORDs
// (pad_amount=3) => LDS row stride 80 B == LDSS bf16 elems.
static __device__ inline void tdm_load_tile(const __bf16* gbase, size_t elem_off,
                                            unsigned lds_off, int ld, int Kdim, int Mdim)
{
    unsigned long long ga = (unsigned long long)(size_t)(const void*)gbase + elem_off * 2ull;
    u32x4 g0;
    g0[0] = 1u;                                          // count=1, user-mode descriptor
    g0[1] = lds_off;                                     // lds_addr (bytes)
    g0[2] = (unsigned)(ga & 0xFFFFFFFFull);              // global_addr[31:0]
    g0[3] = (unsigned)((ga >> 32) & 0x01FFFFFFull)       // global_addr[56:32]
          | (2u << 30);                                  // type = 2 ("image")
    i32x8 g1;
    g1[0] = (1 << 16)                                    // data_size = 2 bytes
          | (1 << 20)                                    // pad_enable
          | (3 << 22)                                    // pad_interval: every 64B
          | (3 << 25);                                   // pad_amount: 4 DWORDs (16B)
    g1[1] = (Kdim & 0xFFFF) << 16;                       // abar=0 | tensor_dim0[15:0]
    g1[2] = ((Kdim >> 16) & 0xFFFF) | ((Mdim & 0xFFFF) << 16);   // dim0 hi | tensor_dim1 lo
    g1[3] = ((Mdim >> 16) & 0xFFFF) | (TK << 16);        // tensor_dim1 hi | tile_dim0
    g1[4] = 128;                                         // tile_dim1 = 128 rows, tile_dim2 = 0
    g1[5] = ld;                                          // tensor_dim0_stride[31:0] (elems)
    g1[6] = 0;                                           // stride0 hi | stride1 lo
    g1[7] = 0;                                           // stride1 hi
    i32x4 z4 = {0, 0, 0, 0};
    i32x8 z8 = {0, 0, 0, 0, 0, 0, 0, 0};
    __builtin_amdgcn_tensor_load_to_lds(g0, g1, z4, z4, z8, 0);
}
static __device__ inline unsigned lds_off_of(const void* p) {
    return (unsigned)((unsigned long long)(size_t)p & 0xFFFFFFFFull);
}
#endif

__launch_bounds__(256)
__global__ void gemm_bf16_kernel(const __bf16* __restrict__ A, int lda,
                                 const __bf16* __restrict__ W, int ldw,
                                 const float*  __restrict__ bias,
                                 const float*  __restrict__ res32,
                                 float*        __restrict__ out32,
                                 __bf16*       __restrict__ outbf,
                                 int M, int N, int K, int epi)
{
    __shared__ __bf16 sA[2][TM][LDSS];
    __shared__ __bf16 sW[2][TN][LDSS];

    const int tid = threadIdx.x;
    const int wid = tid >> 5;
    const int wy  = wid & 1;      // 2 waves over M (64 each)
    const int wx  = wid >> 1;     // 4 waves over N (32 each)
    const int m0  = blockIdx.y * TM;
    const int n0  = blockIdx.x * TN;

    v8f acc[4][2];
    #pragma unroll
    for (int i = 0; i < 4; ++i)
        #pragma unroll
        for (int j = 0; j < 2; ++j) acc[i][j] = zero8();

#if USE_TDM
    const int steps = K / TK;
    if (wid == 0) {
        tdm_load_tile(A, (size_t)m0 * lda, lds_off_of(&sA[0][0][0]), lda, K, M);
        tdm_load_tile(W, (size_t)n0 * ldw, lds_off_of(&sW[0][0][0]), ldw, K, N);
    }
    for (int s = 0; s < steps; ++s) {
        const int p = s & 1;
        if (wid == 0) __builtin_amdgcn_s_wait_tensorcnt(0);   // buf[p] tiles resident
        __syncthreads();                                       // publish; all done with buf[1-p]
        if (wid == 0 && (s + 1) < steps) {
            const size_t k1 = (size_t)(s + 1) * TK;
            tdm_load_tile(A, (size_t)m0 * lda + k1, lds_off_of(&sA[1 - p][0][0]), lda, K, M);
            tdm_load_tile(W, (size_t)n0 * ldw + k1, lds_off_of(&sW[1 - p][0][0]), ldw, K, N);
        }
        v16bf bf0 = frag_b(&sW[p][wx*32 +  0][0], LDSS);
        v16bf bf1 = frag_b(&sW[p][wx*32 + 16][0], LDSS);
        #pragma unroll
        for (int i = 0; i < 4; ++i) {
            v16bf af = frag_a(&sA[p][wy*64 + i*16][0], LDSS);
            acc[i][0] = wmma_bf16(af, bf0, acc[i][0]);
            acc[i][1] = wmma_bf16(af, bf1, acc[i][1]);
        }
    }
#else
    (void)M;
    const int lrow = tid >> 1;            // 0..127
    const int lkp  = (tid & 1) * 16;      // 0 or 16
    for (int k0 = 0; k0 < K; k0 += TK) {
        const v8bf* ga = (const v8bf*)(A + (size_t)(m0 + lrow) * lda + k0 + lkp);
        v8bf a0 = ga[0], a1 = ga[1];
        const v8bf* gw = (const v8bf*)(W + (size_t)(n0 + lrow) * ldw + k0 + lkp);
        v8bf w0 = gw[0], w1 = gw[1];
        __syncthreads();
        *(v8bf*)&sA[0][lrow][lkp]     = a0;
        *(v8bf*)&sA[0][lrow][lkp + 8] = a1;
        *(v8bf*)&sW[0][lrow][lkp]     = w0;
        *(v8bf*)&sW[0][lrow][lkp + 8] = w1;
        __syncthreads();
        v16bf bf0 = frag_b(&sW[0][wx*32 +  0][0], LDSS);
        v16bf bf1 = frag_b(&sW[0][wx*32 + 16][0], LDSS);
        #pragma unroll
        for (int i = 0; i < 4; ++i) {
            v16bf af = frag_a(&sA[0][wy*64 + i*16][0], LDSS);
            acc[i][0] = wmma_bf16(af, bf0, acc[i][0]);
            acc[i][1] = wmma_bf16(af, bf1, acc[i][1]);
        }
    }
#endif

    const int l  = tid & 31;
    const int cn = l & 15;
    const int rb = (l >> 4) * 8;
    #pragma unroll
    for (int i = 0; i < 4; ++i) {
        #pragma unroll
        for (int j = 0; j < 2; ++j) {
            #pragma unroll
            for (int r = 0; r < 8; ++r) {
                int m = m0 + wy*64 + i*16 + rb + r;
                int n = n0 + wx*32 + j*16 + cn;
                float v = acc[i][j][r] + bias[n];
                if (epi == EPI_F32) {
                    out32[(size_t)m * N + n] = v;
                } else if (epi == EPI_RELU_BF16) {
                    outbf[(size_t)m * N + n] = (__bf16)fmaxf(v, 0.f);
                } else if (epi == EPI_RES) {
                    float t = v + res32[(size_t)m * N + n];
                    out32[(size_t)m * N + n] = t;
                    outbf[(size_t)m * N + n] = (__bf16)t;
                } else { // EPI_LOGITS: token m = s*B + b -> out[b, v=n, s]
                    int b = m % B_;
                    int s = m / B_;
                    out32[((size_t)b * V_ + n) * S_ + s] = v;
                }
            }
        }
    }
}

// ---------------- attention: A[b,h] = sum_s v ⊗ k  (64x64, K=S) ----------------
// vT,kT layouts: [B*H][dh][S]  (contiguous in s)
__launch_bounds__(256)
__global__ void attn_accum_kernel(const __bf16* __restrict__ vT,
                                  const __bf16* __restrict__ kT,
                                  __bf16*       __restrict__ Abf)
{
    const int bh = blockIdx.x;
    const __bf16* vt = vT + (size_t)bh * DH_ * S_;
    const __bf16* kt = kT + (size_t)bh * DH_ * S_;
    const int tid = threadIdx.x, wid = tid >> 5;
    const int wn = wid & 3;       // N subtile (dq)
    const int wk = wid >> 2;      // K half

    v8f acc[4];
    #pragma unroll
    for (int i = 0; i < 4; ++i) acc[i] = zero8();

    const int kbeg = wk * (S_/2), kend = kbeg + (S_/2);
    for (int kk = kbeg; kk < kend; kk += 32) {
        v16bf bfr = frag_b(kt + (size_t)(wn*16) * S_ + kk, S_);
        #pragma unroll
        for (int i = 0; i < 4; ++i) {
            v16bf af = frag_a(vt + (size_t)(i*16) * S_ + kk, S_);
            acc[i] = wmma_bf16(af, bfr, acc[i]);
        }
    }

    __shared__ float red[2][DH_][DH_];   // 32 KB
    const int l = tid & 31, cn = l & 15, rb = (l >> 4) * 8;
    #pragma unroll
    for (int i = 0; i < 4; ++i)
        #pragma unroll
        for (int r = 0; r < 8; ++r)
            red[wk][i*16 + rb + r][wn*16 + cn] = acc[i][r];
    __syncthreads();
    for (int e = tid; e < DH_*DH_; e += 256) {
        float s = red[0][e >> 6][e & 63] + red[1][e >> 6][e & 63];
        Abf[(size_t)bh * DH_ * DH_ + e] = (__bf16)s;   // [dv][dq]
    }
}

// ---------------- attention: O[s,dv] = Q[s,:] · A[dv,:]^T ----------------
__launch_bounds__(256)
__global__ void attn_apply_kernel(const __bf16* __restrict__ qbf,   // [T, D]
                                  const __bf16* __restrict__ Abf,   // [B*H][dv][dq]
                                  __bf16*       __restrict__ obf)   // [T, D]
{
    const int bh = blockIdx.y;
    const int b  = bh / H_, hh = bh % H_;
    const int tid = threadIdx.x, wid = tid >> 5;
    const int s0 = blockIdx.x * 128 + wid * 16;
    const __bf16* Ab = Abf + (size_t)bh * DH_ * DH_;

    v8f acc[4];
    #pragma unroll
    for (int j = 0; j < 4; ++j) acc[j] = zero8();

    #pragma unroll
    for (int kq = 0; kq < DH_; kq += 32) {
        v16bf af = frag_a(qbf + ((size_t)s0 * B_ + b) * D_ + hh*DH_ + kq, B_*D_);
        #pragma unroll
        for (int j = 0; j < 4; ++j) {
            v16bf bfr = frag_b(Ab + (size_t)(j*16) * DH_ + kq, DH_);
            acc[j] = wmma_bf16(af, bfr, acc[j]);
        }
    }

    const int l = tid & 31, cn = l & 15, rb = (l >> 4) * 8;
    #pragma unroll
    for (int j = 0; j < 4; ++j)
        #pragma unroll
        for (int r = 0; r < 8; ++r) {
            int s  = s0 + rb + r;
            int dv = j*16 + cn;
            obf[((size_t)s * B_ + b) * D_ + hh*DH_ + dv] = (__bf16)acc[j][r];
        }
}

// ---------------- per-head q/k/v normalize (wave32 per (t,h)) ----------------
__launch_bounds__(256)
__global__ void qkv_norm_kernel(const float* __restrict__ qpre,
                                const float* __restrict__ kpre,
                                const float* __restrict__ vpre,
                                __bf16* __restrict__ qbf,
                                __bf16* __restrict__ kT,
                                __bf16* __restrict__ vT)
{
    const int tid = threadIdx.x, wid = tid >> 5, lane = tid & 31;
    const int idx = blockIdx.x * 8 + wid;      // (t, h)
    const int t = idx / H_, hh = idx % H_;
    const int b = t % B_,  s = t / B_;
    const size_t base = (size_t)t * D_ + hh * DH_;

    // q: unit
    float q0 = qpre[base + lane], q1 = qpre[base + 32 + lane];
    float ss = q0*q0 + q1*q1;
    #pragma unroll
    for (int m = 16; m >= 1; m >>= 1) ss += __shfl_xor(ss, m, 32);
    float rn = rsqrtf(ss);
    qbf[base + lane]      = (__bf16)(q0 * rn);
    qbf[base + 32 + lane] = (__bf16)(q1 * rn);

    // k: elu(unit), stored transposed [bh][dq][s]
    float k0 = kpre[base + lane], k1 = kpre[base + 32 + lane];
    ss = k0*k0 + k1*k1;
    #pragma unroll
    for (int m = 16; m >= 1; m >>= 1) ss += __shfl_xor(ss, m, 32);
    rn = rsqrtf(ss);
    k0 *= rn; k1 *= rn;
    k0 = (k0 > 0.f) ? k0 : (__expf(k0) - 1.f);
    k1 = (k1 > 0.f) ? k1 : (__expf(k1) - 1.f);
    const size_t tb = ((size_t)(b*H_ + hh) * DH_) * S_ + s;
    kT[tb + (size_t)lane * S_]        = (__bf16)k0;
    kT[tb + (size_t)(lane + 32) * S_] = (__bf16)k1;

    // v: straight transpose
    float v0 = vpre[base + lane], v1 = vpre[base + 32 + lane];
    vT[tb + (size_t)lane * S_]        = (__bf16)v0;
    vT[tb + (size_t)(lane + 32) * S_] = (__bf16)v1;
}

// ---------------- LayerNorm (wave32 per token, in-place) ----------------
__launch_bounds__(256)
__global__ void ln_kernel(float* __restrict__ h32, __bf16* __restrict__ hbf,
                          const float* __restrict__ g, const float* __restrict__ be)
{
    const int tid = threadIdx.x, wid = tid >> 5, lane = tid & 31;
    const int t = blockIdx.x * 8 + wid;
    const size_t base = (size_t)t * D_;
    float x[16], sum = 0.f, sq = 0.f;
    #pragma unroll
    for (int i = 0; i < 16; ++i) {
        x[i] = h32[base + i*32 + lane];
        sum += x[i]; sq += x[i]*x[i];
    }
    #pragma unroll
    for (int m = 16; m >= 1; m >>= 1) {
        sum += __shfl_xor(sum, m, 32);
        sq  += __shfl_xor(sq,  m, 32);
    }
    const float mean = sum * (1.f / D_);
    const float var  = sq * (1.f / D_) - mean * mean;
    const float rstd = rsqrtf(var + EPS_);
    #pragma unroll
    for (int i = 0; i < 16; ++i) {
        int d = i*32 + lane;
        float y = (x[i] - mean) * rstd * g[d] + be[d];
        h32[base + d] = y;
        hbf[base + d] = (__bf16)y;
    }
}

// ---------------- embedding gather + bf16 shadow ----------------
__global__ void embed_kernel(const int* __restrict__ inp, const float* __restrict__ emb,
                             float* __restrict__ h32, __bf16* __restrict__ hbf)
{
    size_t i = (size_t)blockIdx.x * blockDim.x + threadIdx.x;   // over T*D
    int t = (int)(i / D_), d = (int)(i % D_);
    int b = t % B_, s = t / B_;
    int id = inp[(size_t)b * S_ + s];
    float v = emb[(size_t)id * D_ + d];
    h32[i] = v;
    hbf[i] = (__bf16)v;
}

// ---------------- fp32 -> bf16 weight conversion ----------------
__global__ void cvt_kernel(const float* __restrict__ src, __bf16* __restrict__ dst, size_t n)
{
    size_t i = (size_t)blockIdx.x * blockDim.x + threadIdx.x;
    if (i < n) dst[i] = (__bf16)src[i];
}

// ---------------- host ----------------
extern "C" void kernel_launch(void* const* d_in, const int* in_sizes, int n_in,
                              void* d_out, int out_size, void* d_ws, size_t ws_size,
                              hipStream_t stream)
{
    (void)in_sizes; (void)n_in; (void)out_size; (void)ws_size;
    const int*   input = (const int*)  d_in[0];
    const float* emb   = (const float*)d_in[1];
    const float* Wq    = (const float*)d_in[2];
    const float* bq    = (const float*)d_in[3];
    const float* Wk    = (const float*)d_in[4];
    const float* bk    = (const float*)d_in[5];
    const float* Wv    = (const float*)d_in[6];
    const float* bv    = (const float*)d_in[7];
    const float* Wo    = (const float*)d_in[8];
    const float* bo    = (const float*)d_in[9];
    const float* W1    = (const float*)d_in[10];
    const float* b1    = (const float*)d_in[11];
    const float* W2    = (const float*)d_in[12];
    const float* b2    = (const float*)d_in[13];
    const float* g2    = (const float*)d_in[14];
    const float* be2   = (const float*)d_in[15];
    const float* Wfc   = (const float*)d_in[16];
    const float* bfc   = (const float*)d_in[17];

    char* ws = (char*)d_ws;
    size_t off = 0;
    auto alloc = [&](size_t bytes) -> char* {
        off = (off + 255) & ~(size_t)255;
        char* p = ws + off;
        off += bytes;
        return p;
    };

    const size_t nQKVO = (size_t)L_ * D_ * D_;
    const size_t nW1   = (size_t)L_ * FF_ * D_;
    const size_t nW2   = (size_t)L_ * D_ * FF_;
    const size_t nFc   = (size_t)V_ * D_;
    const size_t nTD   = (size_t)T_ * D_;
    const size_t nTFF  = (size_t)T_ * FF_;

    __bf16* Wq_bf  = (__bf16*)alloc(nQKVO * 2);
    __bf16* Wk_bf  = (__bf16*)alloc(nQKVO * 2);
    __bf16* Wv_bf  = (__bf16*)alloc(nQKVO * 2);
    __bf16* Wo_bf  = (__bf16*)alloc(nQKVO * 2);
    __bf16* W1_bf  = (__bf16*)alloc(nW1 * 2);
    __bf16* W2_bf  = (__bf16*)alloc(nW2 * 2);
    __bf16* Wfc_bf = (__bf16*)alloc(nFc * 2);
    float*  h32    = (float*) alloc(nTD * 4);
    __bf16* hbf    = (__bf16*)alloc(nTD * 2);
    float*  qpre   = (float*) alloc(nTD * 4);
    float*  kpre   = (float*) alloc(nTD * 4);
    float*  vpre   = (float*) alloc(nTD * 4);
    __bf16* qbf    = (__bf16*)alloc(nTD * 2);
    __bf16* kTb    = (__bf16*)alloc(nTD * 2);   // [B*H][dh][S]
    __bf16* vTb    = (__bf16*)alloc(nTD * 2);
    __bf16* Abf    = (__bf16*)alloc((size_t)B_ * H_ * DH_ * DH_ * 2);
    __bf16* obf    = (__bf16*)alloc(nTD * 2);
    __bf16* f1bf   = (__bf16*)alloc(nTFF * 2);

    auto cvt = [&](const float* s, __bf16* d, size_t n) {
        cvt_kernel<<<(unsigned)((n + 255) / 256), 256, 0, stream>>>(s, d, n);
    };
    cvt(Wq, Wq_bf, nQKVO);  cvt(Wk, Wk_bf, nQKVO);
    cvt(Wv, Wv_bf, nQKVO);  cvt(Wo, Wo_bf, nQKVO);
    cvt(W1, W1_bf, nW1);    cvt(W2, W2_bf, nW2);
    cvt(Wfc, Wfc_bf, nFc);

    embed_kernel<<<(unsigned)(nTD / 256), 256, 0, stream>>>(input, emb, h32, hbf);

    auto gemm = [&](const __bf16* A, int lda, const __bf16* Wb, int ldw,
                    const float* bias, const float* res, float* o32, __bf16* obf_,
                    int M, int N, int K, int epi) {
        dim3 grid(N / TN, M / TM);
        gemm_bf16_kernel<<<grid, 256, 0, stream>>>(A, lda, Wb, ldw, bias, res, o32, obf_, M, N, K, epi);
    };

    for (int l = 0; l < L_; ++l) {
        const __bf16* wq = Wq_bf + (size_t)l * D_ * D_;
        const __bf16* wk = Wk_bf + (size_t)l * D_ * D_;
        const __bf16* wv = Wv_bf + (size_t)l * D_ * D_;
        const __bf16* wo = Wo_bf + (size_t)l * D_ * D_;
        const __bf16* w1 = W1_bf + (size_t)l * FF_ * D_;
        const __bf16* w2 = W2_bf + (size_t)l * D_ * FF_;

        gemm(hbf, D_, wq, D_, bq + (size_t)l*D_, nullptr, qpre, nullptr, T_, D_, D_, EPI_F32);
        gemm(hbf, D_, wk, D_, bk + (size_t)l*D_, nullptr, kpre, nullptr, T_, D_, D_, EPI_F32);
        gemm(hbf, D_, wv, D_, bv + (size_t)l*D_, nullptr, vpre, nullptr, T_, D_, D_, EPI_F32);

        qkv_norm_kernel<<<(T_ * H_) / 8, 256, 0, stream>>>(qpre, kpre, vpre, qbf, kTb, vTb);
        attn_accum_kernel<<<B_ * H_, 256, 0, stream>>>(vTb, kTb, Abf);
        attn_apply_kernel<<<dim3(S_ / 128, B_ * H_), 256, 0, stream>>>(qbf, Abf, obf);

        gemm(obf, D_, wo, D_, bo + (size_t)l*D_, h32, h32, hbf, T_, D_, D_, EPI_RES);
        gemm(hbf, D_, w1, D_, b1 + (size_t)l*FF_, nullptr, nullptr, f1bf, T_, FF_, D_, EPI_RELU_BF16);
        gemm(f1bf, FF_, w2, FF_, b2 + (size_t)l*D_, h32, h32, hbf, T_, D_, FF_, EPI_RES);
        ln_kernel<<<T_ / 8, 256, 0, stream>>>(h32, hbf, g2 + (size_t)l*D_, be2 + (size_t)l*D_);
    }

    gemm(hbf, D_, Wfc_bf, D_, bfc, nullptr, (float*)d_out, nullptr, T_, V_, D_, EPI_LOGITS);
}
